// SwinTransformerBlock_16801912062833
// MI455X (gfx1250) — matrix operand order repaired and checked by reading, not compiled, not run on hardware
//
#include <hip/hip_runtime.h>
#include <math.h>

// ---------------------------------------------------------------------------
// Swin transformer block for MI455X (gfx1250), all GEMMs via
// v_wmma_f32_16x16x32_f16 (f16 inputs, fp32 accumulate).
// Pipeline: pack weights -> LN1 -> QKV -> windowed attention -> proj+res
//           -> LN2 -> FC1+GELU -> FC2+res -> d_out (fp32)
// CDNA5 features: WMMA everywhere, TDM tensor_load_to_lds (with HW pad) for
// the attention V tile, global_prefetch in the GEMM K-loop.
// This round: force x2 unroll of the GEMM K-loop (restores clause/wmma
// density lost to the prefetch code) and deepen/widen the prefetch.
// ---------------------------------------------------------------------------

#define DIM    512
#define HEADS  8
#define HD     64
#define WIN    64
#define HIDDEN 2048
#define TOK    32768            // 8 * 4096 tokens
#define EPS    1e-5f
#define LDST   72               // padded LDS row stride (f16 elems), 16B aligned
#define LDS_V_BYTES (WIN * LDST * 2)      // 9216
#define LDS_TOTAL   (2 * WIN * LDST * 2)  // 18432 (V + P)

typedef __attribute__((ext_vector_type(16))) _Float16 v16h;
typedef __attribute__((ext_vector_type(8)))  _Float16 v8h;
typedef __attribute__((ext_vector_type(8)))  float    v8f;
typedef __attribute__((ext_vector_type(4)))  unsigned int v4u;
typedef __attribute__((ext_vector_type(8)))  int      v8i;
typedef __attribute__((ext_vector_type(4)))  int      v4i;

static __device__ __forceinline__ v8f wmma_16x16x32(v16h a, v16h b, v8f c) {
  return __builtin_amdgcn_wmma_f32_16x16x32_f16(
      /*neg_a=*/false, a, /*neg_b=*/false, b,
      /*c_mod=*/(short)0, c, /*reuse_a=*/false, /*reuse_b=*/false);
}

// A fragment: 16x32 f16 tile from a row-major matrix (lda = elements).
// Contiguous K order (B loads match): lane l: row l&15, K = {kg*16..kg*16+15}, kg = l>>4.
static __device__ __forceinline__ v16h load_a(const _Float16* tile, int lda, int lane) {
  const _Float16* p = tile + (size_t)(lane & 15) * lda + ((lane >> 4) << 4);
  v8h c0 = *(const v8h*)p;
  v8h c1 = *(const v8h*)(p + 8);
  v16h r;
#pragma unroll
  for (int i = 0; i < 8; ++i) { r[i] = c0[i]; r[i + 8] = c1[i]; }
  return r;
}

// B fragment where B = M^T, M row-major (used for Q*K^T): lane n holds column
// n of B = row n of M; K run of 16 is contiguous in M's row -> one 32B load.
static __device__ __forceinline__ v16h load_bt(const _Float16* tile, int ldm, int lane) {
  return *(const v16h*)(tile + (size_t)(lane & 15) * ldm + ((lane >> 4) << 4));
}

// B fragment from row-major B (strided gather; used from LDS for P*V).
static __device__ __forceinline__ v16h load_bg(const _Float16* tile, int ldb, int lane) {
  const _Float16* p = tile + (size_t)((lane >> 4) << 4) * ldb + (lane & 15);
  v16h r;
#pragma unroll
  for (int t = 0; t < 16; ++t) r[t] = p[(size_t)t * ldb];
  return r;
}

// ---------------------------------------------------------------------------
// Weight repack: fp32 W[K x N] row-major -> f16 B-fragment tiles.
// Tile (kb, nt) = 32x16; 32 lanes x 16 contiguous f16 each.
// ---------------------------------------------------------------------------
__global__ void __launch_bounds__(256)
k_pack(const float* __restrict__ W, _Float16* __restrict__ Wp, int K, int N) {
  int idx   = blockIdx.x * blockDim.x + threadIdx.x;
  int total = (K >> 5) * (N >> 4) * 32;
  if (idx >= total) return;
  int lane = idx & 31;
  int tile = idx >> 5;
  int nt   = tile % (N >> 4);
  int kb   = tile / (N >> 4);
  int col  = (nt << 4) + (lane & 15);
  int k0   = (kb << 5) + ((lane >> 4) << 4);
  _Float16* out = Wp + ((size_t)tile << 9) + (lane << 4);
#pragma unroll
  for (int t = 0; t < 16; ++t) out[t] = (_Float16)W[(size_t)(k0 + t) * N + col];
}

// ---------------------------------------------------------------------------
// LayerNorm over C=512, fp32 in -> f16 out. One wave32 per row.
// ---------------------------------------------------------------------------
__global__ void __launch_bounds__(256)
k_ln(const float* __restrict__ x, const float* __restrict__ g,
     const float* __restrict__ b, _Float16* __restrict__ h, int T) {
  int row  = (blockIdx.x * blockDim.x + threadIdx.x) >> 5;
  int lane = threadIdx.x & 31;
  if (row >= T) return;
  const float* xr = x + (size_t)row * DIM;
  float v[16];
  float s = 0.f;
#pragma unroll
  for (int i = 0; i < 16; ++i) { v[i] = xr[lane * 16 + i]; s += v[i]; }
#pragma unroll
  for (int o = 16; o > 0; o >>= 1) s += __shfl_xor(s, o, 32);
  float mu = s * (1.f / DIM);
  float vs = 0.f;
#pragma unroll
  for (int i = 0; i < 16; ++i) { float d = v[i] - mu; vs += d * d; }
#pragma unroll
  for (int o = 16; o > 0; o >>= 1) vs += __shfl_xor(vs, o, 32);
  float rs = rsqrtf(vs * (1.f / DIM) + EPS);
  _Float16* hr = h + (size_t)row * DIM;
#pragma unroll
  for (int i = 0; i < 16; ++i) {
    int c = lane * 16 + i;
    hr[c] = (_Float16)((v[i] - mu) * rs * g[c] + b[c]);
  }
}

// ---------------------------------------------------------------------------
// Generic GEMM: C[T x N] = A[T x K] (f16, row-major) * Bp (packed f16) + bias.
// One wave per 16x64 output strip; 4 WMMAs per 32-deep K step, K-loop
// unrolled x2 so the scheduler can clause two steps of b128 loads.
// MODE 0: f16 out + bias | 1: f16 out + bias + GELU | 2: fp32 out + bias + res
// ---------------------------------------------------------------------------
template <int MODE>
__global__ void __launch_bounds__(256)
k_gemm(const _Float16* __restrict__ A, const _Float16* __restrict__ Bp,
       const float* __restrict__ bias, const float* __restrict__ res,
       _Float16* __restrict__ outh, float* __restrict__ outf,
       int T, int K, int N) {
  int gw    = (blockIdx.x * blockDim.x + threadIdx.x) >> 5;
  int lane  = threadIdx.x & 31;
  int ncg   = N >> 6;
  int tasks = (T >> 4) * ncg;
  if (gw >= tasks) return;
  int rt = gw / ncg, cg = gw % ncg;
  int r0 = rt << 4, n0 = cg << 6;
  int nt16 = N >> 4, nk = K >> 5;

  v8f acc[4] = {};
  const _Float16* Arow = A + (size_t)r0 * K;
#pragma unroll 2
  for (int kb = 0; kb < nk; ++kb) {
    // Prefetch 2 K-steps ahead: A row chunk (16 rows x 64B, one line per
    // lane&15) and the full 4KB of that step's four packed B tiles
    // (lane*128B spans the 4KB) -> global_prefetch_b8.
    if (kb + 2 < nk) {
      __builtin_prefetch(Arow + (size_t)(lane & 15) * K + ((kb + 2) << 5), 0, 0);
      __builtin_prefetch(
          (const char*)(Bp + (((size_t)(kb + 2) * nt16 + (n0 >> 4)) << 9)) +
              lane * 128, 0, 0);
    }
    v16h af = load_a(Arow + (kb << 5), K, lane);
#pragma unroll
    for (int t = 0; t < 4; ++t) {
      const _Float16* bp =
          Bp + (((size_t)kb * nt16 + ((n0 >> 4) + t)) << 9) + (lane << 4);
      acc[t] = wmma_16x16x32(af, *(const v16h*)bp, acc[t]);
    }
  }
  int hi = lane >> 4, cl = lane & 15;
#pragma unroll
  for (int t = 0; t < 4; ++t) {
    int col  = n0 + (t << 4) + cl;
    float bv = bias[col];
#pragma unroll
    for (int r = 0; r < 8; ++r) {
      int row = r0 + (hi << 3) + r;
      float v = acc[t][r] + bv;
      if (MODE == 1) v = 0.5f * v * (1.0f + erff(v * 0.70710678118f));
      if (MODE == 2)
        outf[(size_t)row * N + col] = v + res[(size_t)row * N + col];
      else
        outh[(size_t)row * N + col] = (_Float16)v;
    }
  }
}

// ---------------------------------------------------------------------------
// Windowed attention: one block (4 waves) per (window, head).
// S = Q K^T * scale  -> row softmax -> O = P V.  All matmuls via WMMA.
// V tile is staged into LDS by the Tensor Data Mover (tensor_load_to_lds)
// with hardware padding (pad every 32 DWORDs by 4 DWORDs => row stride 72
// f16), falling back to a manual vector copy when the builtin is absent.
// Dynamic shared memory is used so ldsV sits at LDS offset 0 for the D#.
// ---------------------------------------------------------------------------
__global__ void __launch_bounds__(128)
k_attn(const _Float16* __restrict__ qkv, _Float16* __restrict__ o) {
  extern __shared__ __align__(32) _Float16 smem[];
  _Float16* ldsV = smem;                    // LDS offset 0, WIN x LDST
  _Float16* ldsP = smem + WIN * LDST;       // LDS offset LDS_V_BYTES

  const int LDQ = 3 * DIM;                  // qkv row stride
  int head = blockIdx.x & (HEADS - 1);
  int win  = blockIdx.x / HEADS;
  int t0   = win * WIN;
  int lane = threadIdx.x & 31;
  int wave = threadIdx.x >> 5;              // 0..3 -> 16-row strip

  const _Float16* qb = qkv + (size_t)t0 * LDQ + head * HD;
  const _Float16* kp = qb + DIM;
  const _Float16* vp = qb + 2 * DIM;

#if __has_builtin(__builtin_amdgcn_tensor_load_to_lds)
  // --- TDM: DMA the 64x64 f16 V tile (tensor row stride 1536) into LDS with
  //     padding: pad_interval=4 (every 32 DWORDs = one 64-elem row),
  //     pad_amount=3 (4 DWORDs = 8 f16) -> effective LDS row stride 72.
  if (threadIdx.x < 32) {                   // one wave issues the descriptor
    unsigned long long ga = (unsigned long long)(uintptr_t)vp;
    v4u g0;
    g0[0] = 1u;                                        // count=1 (valid D#)
    g0[1] = 0u;                                        // lds_addr = 0 (ldsV)
    g0[2] = (unsigned)(ga & 0xffffffffu);              // global_addr[31:0]
    g0[3] = (unsigned)((ga >> 32) & 0x01ffffffu)       // global_addr[56:32]
          | (2u << 30);                                // type = 2 ("image")
    v8i g1;
    g1[0] = (1 << 16)    // data_size = 1 -> 2 bytes
          | (1 << 20)    // pad_enable
          | (4 << 22)    // pad_interval: every 32 DWORDs (= 128B row)
          | (3 << 25);   // pad_amount: 4 DWORDs (= 16B = 8 f16)
    g1[1] = (WIN << 16); // tensor_dim0 = 64 (low 16 bits)
    g1[2] = (WIN << 16); // tensor_dim0 hi = 0 | tensor_dim1 = 64 (low 16)
    g1[3] = (WIN << 16); // tensor_dim1 hi = 0 | tile_dim0 = 64
    g1[4] = WIN;         // tile_dim1 = 64, tile_dim2 = 0 (2D)
    g1[5] = LDQ;         // tensor_dim0_stride = 1536 (low 32)
    g1[6] = 0;
    g1[7] = 0;
    v4i gz = {0, 0, 0, 0};
#if defined(__clang_major__) && (__clang_major__ >= 23)
    v8i gz8 = {0, 0, 0, 0, 0, 0, 0, 0};
    __builtin_amdgcn_tensor_load_to_lds(g0, g1, gz, gz, gz8, 0);
#else
    __builtin_amdgcn_tensor_load_to_lds(g0, g1, gz, gz, 0);
#endif
    __builtin_amdgcn_s_wait_tensorcnt(0);
  }
#else
  // Fallback: manual staged copy, 128 threads, half-row each.
  {
    int row = threadIdx.x >> 1;
    int c0  = (threadIdx.x & 1) << 5;
    const v8h* src = (const v8h*)(vp + (size_t)row * LDQ + c0);
    v8h* dst = (v8h*)(ldsV + row * LDST + c0);
    dst[0] = src[0]; dst[1] = src[1]; dst[2] = src[2]; dst[3] = src[3];
  }
#endif

  // S = Q K^T (K^T fragments are contiguous loads from row-major K).
  v8f acc[4] = {};
  const _Float16* Aq = qb + (size_t)(wave * 16) * LDQ;
#pragma unroll
  for (int kblk = 0; kblk < 2; ++kblk) {
    v16h af = load_a(Aq + kblk * 32, LDQ, lane);
#pragma unroll
    for (int t = 0; t < 4; ++t) {
      v16h bf = load_bt(kp + (size_t)(t * 16) * LDQ + kblk * 32, LDQ, lane);
      acc[t] = wmma_16x16x32(af, bf, acc[t]);
    }
  }

  // Row softmax with scale = 1/sqrt(64).  Row m of the C tile lives in
  // vgpr r = m&7, lane half hi = m>>3; reduce across the 16 lanes of a half.
  const float scale = 0.125f;
  int hi = lane >> 4, cl = lane & 15;
#pragma unroll
  for (int r = 0; r < 8; ++r) {
    float m = -3.4e38f;
#pragma unroll
    for (int t = 0; t < 4; ++t) m = fmaxf(m, acc[t][r]);
#pragma unroll
    for (int s = 1; s < 16; s <<= 1) m = fmaxf(m, __shfl_xor(m, s, 32));
    float e[4], sum = 0.f;
#pragma unroll
    for (int t = 0; t < 4; ++t) { e[t] = __expf((acc[t][r] - m) * scale); sum += e[t]; }
#pragma unroll
    for (int s = 1; s < 16; s <<= 1) sum += __shfl_xor(sum, s, 32);
    float inv = 1.f / sum;
    int row = (wave << 4) + (hi << 3) + r;
#pragma unroll
    for (int t = 0; t < 4; ++t)
      ldsP[row * LDST + (t << 4) + cl] = (_Float16)(e[t] * inv);
  }
  __syncthreads();  // ldsV DMA complete + ldsP written, visible to all waves

  // O = P V
  v8f oc[4] = {};
  const _Float16* Ap = ldsP + (wave * 16) * LDST;
#pragma unroll
  for (int kblk = 0; kblk < 2; ++kblk) {
    v16h af = load_a(Ap + kblk * 32, LDST, lane);
#pragma unroll
    for (int t = 0; t < 4; ++t) {
      v16h bf = load_bg(ldsV + (kblk * 32) * LDST + (t << 4), LDST, lane);
      oc[t] = wmma_16x16x32(af, bf, oc[t]);
    }
  }
  _Float16* ob = o + (size_t)(t0 + wave * 16) * DIM + head * HD;
#pragma unroll
  for (int t = 0; t < 4; ++t)
#pragma unroll
    for (int r = 0; r < 8; ++r)
      ob[(size_t)((hi << 3) + r) * DIM + (t << 4) + cl] = (_Float16)oc[t][r];
}

// ---------------------------------------------------------------------------
extern "C" void kernel_launch(void* const* d_in, const int* in_sizes, int n_in,
                              void* d_out, int out_size, void* d_ws, size_t ws_size,
                              hipStream_t stream) {
  (void)in_sizes; (void)n_in; (void)out_size; (void)ws_size;
  const float* x      = (const float*)d_in[0];
  const float* ln1_g  = (const float*)d_in[1];
  const float* ln1_b  = (const float*)d_in[2];
  const float* qkv_w  = (const float*)d_in[3];
  const float* qkv_b  = (const float*)d_in[4];
  const float* proj_w = (const float*)d_in[5];
  const float* proj_b = (const float*)d_in[6];
  const float* ln2_g  = (const float*)d_in[7];
  const float* ln2_b  = (const float*)d_in[8];
  const float* fc1_w  = (const float*)d_in[9];
  const float* fc1_b  = (const float*)d_in[10];
  const float* fc2_w  = (const float*)d_in[11];
  const float* fc2_b  = (const float*)d_in[12];
  float* out = (float*)d_out;

  // Workspace carve-up (~262 MB), with buffer reuse across dead ranges.
  char* ws = (char*)d_ws;
  size_t off = 0;
  auto alloc = [&](size_t bytes) -> char* {
    char* p = ws + off;
    off += (bytes + 255) & ~(size_t)255;
    return p;
  };
  _Float16* bufA = (_Float16*)alloc((size_t)TOK * HIDDEN * 2); // qkv, then fc1 act
  _Float16* h1   = (_Float16*)alloc((size_t)TOK * DIM * 2);    // LN1 out, then LN2 out
  _Float16* bufO = (_Float16*)alloc((size_t)TOK * DIM * 2);    // attention out
  float*    x1   = (float*)   alloc((size_t)TOK * DIM * 4);    // x + proj residual
  _Float16* WpQ  = (_Float16*)alloc((size_t)DIM * 3 * DIM * 2);
  _Float16* WpP  = (_Float16*)alloc((size_t)DIM * DIM * 2);
  _Float16* Wp1  = (_Float16*)alloc((size_t)DIM * HIDDEN * 2);
  _Float16* Wp2  = (_Float16*)alloc((size_t)HIDDEN * DIM * 2);

  // Pack weights into WMMA B-fragment layout (f16).
  k_pack<<<192, 256, 0, stream>>>(qkv_w, WpQ, DIM, 3 * DIM);
  k_pack<<<64,  256, 0, stream>>>(proj_w, WpP, DIM, DIM);
  k_pack<<<256, 256, 0, stream>>>(fc1_w, Wp1, DIM, HIDDEN);
  k_pack<<<256, 256, 0, stream>>>(fc2_w, Wp2, HIDDEN, DIM);

  // LN1
  k_ln<<<TOK / 8, 256, 0, stream>>>(x, ln1_g, ln1_b, h1, TOK);
  // QKV: (TOK/16)*(1536/64) = 49152 wave tasks -> 6144 blocks
  k_gemm<0><<<6144, 256, 0, stream>>>(h1, WpQ, qkv_b, nullptr, bufA, nullptr,
                                      TOK, DIM, 3 * DIM);
  // Windowed attention: 512 windows x 8 heads, dynamic LDS (V + P, padded)
  k_attn<<<(TOK / WIN) * HEADS, 128, LDS_TOTAL, stream>>>(bufA, bufO);
  // Proj + residual(x) -> x1 (fp32): 16384 tasks -> 2048 blocks
  k_gemm<2><<<2048, 256, 0, stream>>>(bufO, WpP, proj_b, x, nullptr, x1,
                                      TOK, DIM, DIM);
  // LN2 (reuse h1)
  k_ln<<<TOK / 8, 256, 0, stream>>>(x1, ln2_g, ln2_b, h1, TOK);
  // FC1 + exact GELU: 65536 tasks -> 8192 blocks (reuse bufA)
  k_gemm<1><<<8192, 256, 0, stream>>>(h1, Wp1, fc1_b, nullptr, bufA, nullptr,
                                      TOK, DIM, HIDDEN);
  // FC2 + residual(x1) -> out (fp32): 16384 tasks -> 2048 blocks
  k_gemm<2><<<2048, 256, 0, stream>>>(bufA, Wp2, fc2_b, x1, nullptr, out,
                                      TOK, HIDDEN, DIM);
}
